// Model_43954695307870
// MI455X (gfx1250) — matrix operand order repaired and checked by reading, not compiled, not run on hardware
//
#include <hip/hip_runtime.h>
#include <hip/hip_bf16.h>

// ---------------------------------------------------------------------------
// CDNA5 / gfx1250 GraphSAGE pipeline.
// Edge-message and node-update GEMMs run on v_wmma_f32_16x16x32_bf16.
// f32->bf16 via v_cvt_pk_bf16_f32 (__builtin_convertvector).
// Weights pre-packed fragment-ordered: one B fragment = 2x ds_load_b128.
// A fragments built once per 16-row tile, N processed in groups of <=4
// 16x16 tiles. __launch_bounds__(256,4) caps VGPRs for occupancy; full
// tiles take an unguarded scatter/store path (no exec-mask juggling).
// ---------------------------------------------------------------------------

typedef __attribute__((ext_vector_type(16))) __bf16       v16bf;
typedef __attribute__((ext_vector_type(2)))  __bf16       v2bf;
typedef __attribute__((ext_vector_type(8)))  float        v8f;
typedef __attribute__((ext_vector_type(4)))  float        v4f;
typedef __attribute__((ext_vector_type(2)))  float        v2f;
typedef __attribute__((ext_vector_type(8)))  unsigned int v8u;
typedef __attribute__((ext_vector_type(4)))  unsigned int v4u;

__device__ __forceinline__ unsigned cvtpk(float x, float y) {
    v2f f; f[0] = x; f[1] = y;
    v2bf b = __builtin_convertvector(f, v2bf);
    return __builtin_bit_cast(unsigned, b);
}
__device__ __forceinline__ void atomicAddF32(float* p, float v) {
    __hip_atomic_fetch_add(p, v, __ATOMIC_RELAXED, __HIP_MEMORY_SCOPE_AGENT);
}
__device__ __forceinline__ v16bf load_frag_b128x2(const unsigned* base) {
    const v4u* wp = (const v4u*)base;           // 32B aligned by construction
    v4u lo = wp[0], hi = wp[1];
    v8u r;
    r[0] = lo[0]; r[1] = lo[1]; r[2] = lo[2]; r[3] = lo[3];
    r[4] = hi[0]; r[5] = hi[1]; r[6] = hi[2]; r[7] = hi[3];
    return __builtin_bit_cast(v16bf, r);
}

// ---------------------------------------------------------------------------
__global__ __launch_bounds__(256) void zero_f32(float* p, size_t n) {
    size_t i = (size_t)blockIdx.x * blockDim.x + threadIdx.x;
    size_t st = (size_t)gridDim.x * blockDim.x;
    for (; i < n; i += st) p[i] = 0.0f;
}

// Fragment-ordered bf16 weight pack.
// W: (OUT x K) row-major f32.  Element i decomposes as
// i = (((kk*NT + nt)*2 + half)*16 + col)*8 + p ;  k = kk*32 + half*16 + 2p,
// n = nt*16 + col ;  out[i] = {bf16 W[n][k], bf16 W[n][k+1]}.
template <int K, int OUT>
__global__ __launch_bounds__(256) void pack_weight_frag(const float* __restrict__ W,
                                                        unsigned* __restrict__ out) {
    constexpr int NT = OUT / 16;
    const int total = (K / 2) * OUT;
    for (int i = blockIdx.x * blockDim.x + threadIdx.x; i < total;
         i += gridDim.x * blockDim.x) {
        int t    = i;
        int p    = t & 7;  t >>= 3;
        int col  = t & 15; t >>= 4;
        int half = t & 1;  t >>= 1;
        int nt   = t % NT;
        int kk   = t / NT;
        int k = kk * 32 + half * 16 + 2 * p;
        int n = nt * 16 + col;
        out[i] = cvtpk(W[(size_t)n * K + k], W[(size_t)n * K + k + 1]);
    }
}

__global__ __launch_bounds__(256) void degree_kernel(const int* __restrict__ dst,
                                                     float* __restrict__ deg, int E) {
    for (int i = blockIdx.x * blockDim.x + threadIdx.x; i < E;
         i += gridDim.x * blockDim.x)
        atomicAddF32(&deg[dst[i]], 1.0f);
}

// ---------------------------------------------------------------------------
// Edge message GEMM:  msg[16 edges x OUT] = [h[src] | e] @ W^T (+bias),
// scattered with f32 atomics into sums[dst].  K = HDIM + 64.
// ---------------------------------------------------------------------------
template <int HDIM, int OUT>
__global__ __launch_bounds__(256, 4) void edge_msg(
    const float* __restrict__ h, const float* __restrict__ efeat,
    const int* __restrict__ src, const int* __restrict__ dst,
    const unsigned* __restrict__ Wpk, const float* __restrict__ bias,
    float* __restrict__ sums, int E) {
    constexpr int K   = HDIM + 64;
    constexpr int NK  = K / 32;
    constexpr int KP  = K / 2;
    constexpr int NT  = OUT / 16;
    constexpr int NTG = (NT < 4) ? NT : 4;   // n-tiles per accumulator group
    constexpr int NG  = NT / NTG;
    __shared__ unsigned ldsW[KP * OUT];
    for (int i = threadIdx.x; i < KP * OUT; i += blockDim.x) ldsW[i] = Wpk[i];
    __syncthreads();

    const int lane = threadIdx.x & 31;
    const int wave = threadIdx.x >> 5;
    const int col  = lane & 15;
    const int half = lane >> 4;

    float bcache[NT];
#pragma unroll
    for (int nt = 0; nt < NT; ++nt) bcache[nt] = bias[nt * 16 + col];

    const int ntiles = (E + 15) >> 4;
    for (int tile = blockIdx.x * 8 + wave; tile < ntiles; tile += gridDim.x * 8) {
        const int  ebase = tile * 16;
        const bool full  = (ebase + 16 <= E);
        int aedge = ebase + col;
        if (aedge >= E) aedge = E - 1;
        const int    s    = src[aedge];
        const float* hrow = h + (size_t)s * HDIM;
        const float* erow = efeat + (size_t)aedge * 64;

        // ---- build all A fragments once (held in registers) ----
        v16bf afrag[NK];
#pragma unroll
        for (int kk = 0; kk < NK; ++kk) {
            const int    ks0 = kk * 32 + half * 16;
            const float* s0  = (ks0 < HDIM) ? hrow + ks0 : erow + (ks0 - HDIM);
            v4f f0 = *(const v4f*)s0;
            v4f f1 = *(const v4f*)(s0 + 4);
            const int    ks1 = ks0 + 8;
            const float* s1  = (ks1 < HDIM) ? hrow + ks1 : erow + (ks1 - HDIM);
            v4f g0 = *(const v4f*)s1;
            v4f g1 = *(const v4f*)(s1 + 4);
            v8u araw;
            araw[0] = cvtpk(f0[0], f0[1]); araw[1] = cvtpk(f0[2], f0[3]);
            araw[2] = cvtpk(f1[0], f1[1]); araw[3] = cvtpk(f1[2], f1[3]);
            araw[4] = cvtpk(g0[0], g0[1]); araw[5] = cvtpk(g0[2], g0[3]);
            araw[6] = cvtpk(g1[0], g1[1]); araw[7] = cvtpk(g1[2], g1[3]);
            afrag[kk] = __builtin_bit_cast(v16bf, araw);
        }

        int dcache[8];
#pragma unroll
        for (int r = 0; r < 8; ++r) {
            int ed = ebase + half * 8 + r;
            dcache[r] = (ed < E) ? dst[ed] : 0;
        }

        // ---- process N in groups of NTG tiles ----
#pragma unroll
        for (int g = 0; g < NG; ++g) {
            v8f acc[NTG] = {};
#pragma unroll
            for (int kk = 0; kk < NK; ++kk) {
#pragma unroll
                for (int j = 0; j < NTG; ++j) {
                    const int nt = g * NTG + j;
                    v16bf b = load_frag_b128x2(
                        ldsW + ((((kk * NT + nt) * 2 + half) * 16 + col) * 8));
                    acc[j] = __builtin_amdgcn_wmma_f32_16x16x32_bf16(
                        false, afrag[kk], false, b, (short)0, acc[j], false, false);
                }
            }
            // scatter: C/D layout -> VGPR r holds M = r + 8*half, N = col
            if (full) {
#pragma unroll
                for (int j = 0; j < NTG; ++j) {
                    const int   nt = g * NTG + j;
                    const int   n  = nt * 16 + col;
                    const float bn = bcache[nt];
#pragma unroll
                    for (int r = 0; r < 8; ++r)
                        atomicAddF32(&sums[(size_t)dcache[r] * OUT + n],
                                     acc[j][r] + bn);
                }
            } else {
#pragma unroll
                for (int j = 0; j < NTG; ++j) {
                    const int   nt = g * NTG + j;
                    const int   n  = nt * 16 + col;
                    const float bn = bcache[nt];
#pragma unroll
                    for (int r = 0; r < 8; ++r)
                        if (ebase + half * 8 + r < E)
                            atomicAddF32(&sums[(size_t)dcache[r] * OUT + n],
                                         acc[j][r] + bn);
                }
            }
        }
    }
}

// ---------------------------------------------------------------------------
// Node update GEMM: hout = relu([hprev | sums/deg] @ W^T + bias)
// ---------------------------------------------------------------------------
template <int HPREV, int AGG, int OUT>
__global__ __launch_bounds__(256, 4) void node_update(
    const float* __restrict__ hprev, const float* __restrict__ sums,
    const float* __restrict__ deg, const unsigned* __restrict__ Wpk,
    const float* __restrict__ bias, float* __restrict__ hout, int N) {
    constexpr int K   = HPREV + AGG;
    constexpr int NK  = K / 32;
    constexpr int KP  = K / 2;
    constexpr int NT  = OUT / 16;
    constexpr int NTG = (NT < 4) ? NT : 4;
    constexpr int NG  = NT / NTG;
    __shared__ unsigned ldsW[KP * OUT];
    for (int i = threadIdx.x; i < KP * OUT; i += blockDim.x) ldsW[i] = Wpk[i];
    __syncthreads();

    const int lane = threadIdx.x & 31;
    const int wave = threadIdx.x >> 5;
    const int col  = lane & 15;
    const int half = lane >> 4;

    float bcache[NT];
#pragma unroll
    for (int nt = 0; nt < NT; ++nt) bcache[nt] = bias[nt * 16 + col];

    const int ntiles = (N + 15) >> 4;
    for (int tile = blockIdx.x * 8 + wave; tile < ntiles; tile += gridDim.x * 8) {
        const int  nbase = tile * 16;
        const bool full  = (nbase + 16 <= N);
        int anode = nbase + col;
        if (anode >= N) anode = N - 1;
        const float* hrow = hprev + (size_t)anode * HPREV;
        const float* srow = sums + (size_t)anode * AGG;
        const float  inv  = 1.0f / fmaxf(deg[anode], 1.0f);

        // ---- build all A fragments once ----
        v16bf afrag[NK];
#pragma unroll
        for (int kk = 0; kk < NK; ++kk) {
            const int ks0 = kk * 32 + half * 16;
            v4f f0, f1;
            if (ks0 < HPREV) {
                f0 = *(const v4f*)(hrow + ks0);
                f1 = *(const v4f*)(hrow + ks0 + 4);
            } else {
                f0 = *(const v4f*)(srow + (ks0 - HPREV)) * inv;
                f1 = *(const v4f*)(srow + (ks0 - HPREV) + 4) * inv;
            }
            const int ks1 = ks0 + 8;
            v4f g0, g1;
            if (ks1 < HPREV) {
                g0 = *(const v4f*)(hrow + ks1);
                g1 = *(const v4f*)(hrow + ks1 + 4);
            } else {
                g0 = *(const v4f*)(srow + (ks1 - HPREV)) * inv;
                g1 = *(const v4f*)(srow + (ks1 - HPREV) + 4) * inv;
            }
            v8u araw;
            araw[0] = cvtpk(f0[0], f0[1]); araw[1] = cvtpk(f0[2], f0[3]);
            araw[2] = cvtpk(f1[0], f1[1]); araw[3] = cvtpk(f1[2], f1[3]);
            araw[4] = cvtpk(g0[0], g0[1]); araw[5] = cvtpk(g0[2], g0[3]);
            araw[6] = cvtpk(g1[0], g1[1]); araw[7] = cvtpk(g1[2], g1[3]);
            afrag[kk] = __builtin_bit_cast(v16bf, araw);
        }

        // ---- N groups ----
#pragma unroll
        for (int g = 0; g < NG; ++g) {
            v8f acc[NTG] = {};
#pragma unroll
            for (int kk = 0; kk < NK; ++kk) {
#pragma unroll
                for (int j = 0; j < NTG; ++j) {
                    const int nt = g * NTG + j;
                    v16bf b = load_frag_b128x2(
                        ldsW + ((((kk * NT + nt) * 2 + half) * 16 + col) * 8));
                    acc[j] = __builtin_amdgcn_wmma_f32_16x16x32_bf16(
                        false, afrag[kk], false, b, (short)0, acc[j], false, false);
                }
            }
            if (full) {
#pragma unroll
                for (int j = 0; j < NTG; ++j) {
                    const int   nt = g * NTG + j;
                    const int   n  = nt * 16 + col;
                    const float bn = bcache[nt];
#pragma unroll
                    for (int r = 0; r < 8; ++r) {
                        const int node = nbase + half * 8 + r;
                        hout[(size_t)node * OUT + n] = fmaxf(acc[j][r] + bn, 0.0f);
                    }
                }
            } else {
#pragma unroll
                for (int j = 0; j < NTG; ++j) {
                    const int   nt = g * NTG + j;
                    const int   n  = nt * 16 + col;
                    const float bn = bcache[nt];
#pragma unroll
                    for (int r = 0; r < 8; ++r) {
                        const int node = nbase + half * 8 + r;
                        if (node < N)
                            hout[(size_t)node * OUT + n] = fmaxf(acc[j][r] + bn, 0.0f);
                    }
                }
            }
        }
    }
}

// ---------------------------------------------------------------------------
// score = [h2[src] | h2[dst]] @ Wp^T + bp   (OUT=2: scalar path)
// ---------------------------------------------------------------------------
__global__ __launch_bounds__(256) void predictor(
    const float* __restrict__ h, const int* __restrict__ src,
    const int* __restrict__ dst, const float* __restrict__ Wp,
    const float* __restrict__ bp, float* __restrict__ out, int E) {
    __shared__ float w[256];   // Wp is 2 x 128
    for (int i = threadIdx.x; i < 256; i += blockDim.x) w[i] = Wp[i];
    __syncthreads();
    const float b0 = bp[0], b1 = bp[1];
    for (int e = blockIdx.x * blockDim.x + threadIdx.x; e < E;
         e += gridDim.x * blockDim.x) {
        const float* hs = h + (size_t)src[e] * 64;
        const float* hd = h + (size_t)dst[e] * 64;
        float s0 = b0, s1 = b1;
#pragma unroll 8
        for (int k = 0; k < 64; ++k) {
            float a = hs[k], c = hd[k];
            s0 += a * w[k]       + c * w[64 + k];
            s1 += a * w[128 + k] + c * w[192 + k];
        }
        out[(size_t)e * 2 + 0] = s0;
        out[(size_t)e * 2 + 1] = s1;
    }
}

// ---------------------------------------------------------------------------
extern "C" void kernel_launch(void* const* d_in, const int* in_sizes, int n_in,
                              void* d_out, int out_size, void* d_ws, size_t ws_size,
                              hipStream_t stream) {
    const float* nfeats = (const float*)d_in[0];
    const float* efeats = (const float*)d_in[1];
    const int*   src    = (const int*)d_in[2];
    const int*   dst    = (const int*)d_in[3];
    const float* Wm1    = (const float*)d_in[4];
    const float* bm1    = (const float*)d_in[5];
    const float* Wa1    = (const float*)d_in[6];
    const float* ba1    = (const float*)d_in[7];
    const float* Wm2    = (const float*)d_in[8];
    const float* bm2    = (const float*)d_in[9];
    const float* Wa2    = (const float*)d_in[10];
    const float* ba2    = (const float*)d_in[11];
    const float* Wp     = (const float*)d_in[12];
    const float* bp     = (const float*)d_in[13];
    float*       out    = (float*)d_out;

    const int N = in_sizes[0] / 64;   // nfeats: N x 1 x 64
    const int E = in_sizes[2];        // src:    E

    char*  ws  = (char*)d_ws;
    size_t off = 0;
    auto alloc = [&](size_t bytes) -> void* {
        void* p = ws + off;
        off = (off + bytes + 255) & ~(size_t)255;
        return p;
    };
    float*    deg   = (float*)alloc((size_t)N * 4);
    float*    sums1 = (float*)alloc((size_t)N * 128 * 4);
    float*    sums2 = (float*)alloc((size_t)N * 64 * 4);
    float*    h1    = (float*)alloc((size_t)N * 128 * 4);
    float*    h2    = (float*)alloc((size_t)N * 64 * 4);
    unsigned* Wm1p  = (unsigned*)alloc(64 * 128 * 4);
    unsigned* Wa1p  = (unsigned*)alloc(96 * 128 * 4);
    unsigned* Wm2p  = (unsigned*)alloc(96 * 64 * 4);
    unsigned* Wa2p  = (unsigned*)alloc(96 * 64 * 4);

    // zero deg + sums1 + sums2 (contiguous region incl. padding)
    size_t zcount = (size_t)((char*)sums2 - (char*)deg) / 4 + (size_t)N * 64;
    zero_f32<<<2048, 256, 0, stream>>>(deg, zcount);

    pack_weight_frag<128, 128><<<64, 256, 0, stream>>>(Wm1, Wm1p);
    pack_weight_frag<192, 128><<<64, 256, 0, stream>>>(Wa1, Wa1p);
    pack_weight_frag<192, 64><<<64, 256, 0, stream>>>(Wm2, Wm2p);
    pack_weight_frag<192, 64><<<64, 256, 0, stream>>>(Wa2, Wa2p);

    degree_kernel<<<1024, 256, 0, stream>>>(dst, deg, E);

    // layer 1
    edge_msg<64, 128><<<2048, 256, 0, stream>>>(nfeats, efeats, src, dst, Wm1p, bm1, sums1, E);
    node_update<64, 128, 128><<<512, 256, 0, stream>>>(nfeats, sums1, deg, Wa1p, ba1, h1, N);
    // layer 2
    edge_msg<128, 64><<<2048, 256, 0, stream>>>(h1, efeats, src, dst, Wm2p, bm2, sums2, E);
    node_update<128, 64, 64><<<512, 256, 0, stream>>>(h1, sums2, deg, Wa2p, ba2, h2, N);
    // predictor
    predictor<<<3125, 256, 0, stream>>>(h2, src, dst, Wp, bp, out, E);
}
